// MultiHeadAttention_8718783611695
// MI455X (gfx1250) — compile-verified, matrix operand order checked by reading, not run on hardware
//
#include <hip/hip_runtime.h>
#include <hip/hip_bf16.h>
#include <stdint.h>

// ---------------------------------------------------------------------------
// MHA forward for B=4,T=1024,C=1024,H=16,D=64 (fp32 in/out) on gfx1250.
// GEMMs via v_wmma_f32_16x16x32_bf16; tile staging via async global->LDS
// (ASYNCcnt pipeline) with double buffering.
// ---------------------------------------------------------------------------

typedef __attribute__((ext_vector_type(16))) __bf16 v16bf;
typedef __attribute__((ext_vector_type(8)))  __bf16 v8bf;
typedef __attribute__((ext_vector_type(8)))  float  v8f;
typedef __attribute__((ext_vector_type(4)))  int    v4i;

#define BB 4
#define TT 1024
#define CC 1024
#define HH 16
#define DD 64

#if __has_builtin(__builtin_amdgcn_global_load_async_to_lds_b128)
#define HAVE_ASYNC_LDS 1
#else
#define HAVE_ASYNC_LDS 0
#endif

__device__ __forceinline__ __bf16 f2bf(float f) {
  union { float f; uint32_t u; } x; x.f = f;
  uint32_t r = x.u + 0x7FFFu + ((x.u >> 16) & 1u);   // round-to-nearest-even
  union { uint16_t u; __bf16 b; } y; y.u = (uint16_t)(r >> 16);
  return y.b;
}
__device__ __forceinline__ __bf16 to_bf(float f)  { return f2bf(f); }
__device__ __forceinline__ __bf16 to_bf(__bf16 b) { return b; }

__device__ __forceinline__ v8f wmma_bf16(v16bf a, v16bf b, v8f c) {
  // (neg_a, A, neg_b, B, c_mod, C, reuse_a, reuse_b)
  return __builtin_amdgcn_wmma_f32_16x16x32_bf16(false, a, false, b,
                                                 (short)0, c, false, false);
}

// 16-byte global -> LDS copy; async (ASYNCcnt) when available.
__device__ __forceinline__ void copy16_g2l(__bf16* lds, const __bf16* g) {
#if HAVE_ASYNC_LDS
  __builtin_amdgcn_global_load_async_to_lds_b128(
      (__attribute__((address_space(1))) v4i*)g,
      (__attribute__((address_space(3))) v4i*)lds, 0, 0);
#else
  *(v8bf*)lds = *(const v8bf*)g;
#endif
}

template<int N>
__device__ __forceinline__ void async_wait_le() {
#if HAVE_ASYNC_LDS
#if __has_builtin(__builtin_amdgcn_s_wait_asynccnt)
  __builtin_amdgcn_s_wait_asynccnt(N);
#else
  asm volatile("s_wait_asynccnt %0" :: "i"(N) : "memory");
#endif
#endif
}

// Load a 16x32 A-style fragment from a row-major bf16 matrix (LDS or global).
// Lane L: row = row0 + (L&15), K chunks at (L>=16 ? 8 : 0) and +16,
// each 8 contiguous bf16 (16 bytes).
__device__ __forceinline__ v16bf load_frag_rm(const __bf16* base, int row0, int ld) {
  int lane = threadIdx.x & 31;
  const __bf16* p = base + (size_t)(row0 + (lane & 15)) * ld + ((lane >> 4) << 3);
  union { v16bf v; v8bf h[2]; } u;
  u.h[0] = *(const v8bf*)(p);
  u.h[1] = *(const v8bf*)(p + 16);
  return u.v;
}

// ---------------------------------------------------------------------------
// Elementwise fp32 -> bf16
// ---------------------------------------------------------------------------
__global__ void cvt_f32_bf16_kernel(const float* __restrict__ in,
                                    __bf16* __restrict__ out, int n) {
  int i = blockIdx.x * blockDim.x + threadIdx.x;
  if (i < n) out[i] = f2bf(in[i]);
}

// Batched transpose (+convert): out[g][c][r] = in[g][r][c]
template<typename T>
__global__ void transpose_bf16_kernel(const T* __restrict__ in,
                                      __bf16* __restrict__ out,
                                      int G, int R, int C) {
  long long i = (long long)blockIdx.x * blockDim.x + threadIdx.x;
  long long total = (long long)G * R * C;
  if (i >= total) return;
  int c = (int)(i % C);
  long long t = i / C;
  int r = (int)(t % R);
  int g = (int)(t / R);
  out[((long long)g * C + c) * R + r] = to_bf(in[i]);
}

// ---------------------------------------------------------------------------
// GEMM: Out[M,N] = A[M,K] * B[K,N], with B given as Bt[N,K] (row-major).
// Workgroup = 128 threads (4 waves), tile 64x64, K-step 32.
// Double-buffered LDS, staged with async global->LDS (ASYNCcnt pipeline).
// FINAL: f32 output + bias; else bf16 output.
// ---------------------------------------------------------------------------
template<bool FINAL>
__global__ __launch_bounds__(128)
void gemm_bt_kernel(const __bf16* __restrict__ A,
                    const __bf16* __restrict__ Bt,
                    __bf16* __restrict__ Obf,
                    float* __restrict__ Of,
                    const float* __restrict__ bias,
                    int M, int N, int K) {
  __shared__ alignas(16) __bf16 As[2][64 * 32];
  __shared__ alignas(16) __bf16 Bs[2][64 * 32];
  const int tid  = threadIdx.x;
  const int lane = tid & 31;
  const int wave = tid >> 5;
  const int wm = wave & 1, wn = wave >> 1;
  const int m0 = blockIdx.y * 64;
  const int n0 = blockIdx.x * 64;

  // Stage one 64x32 tile of A and of Bt into LDS buffer `buf`.
  // 256 16-byte chunks per matrix, 2 per thread -> 4 async ops / thread.
  auto stage = [&](int buf, int k0) {
#pragma unroll
    for (int i = 0; i < 2; ++i) {
      int c   = tid + i * 128;
      int row = c >> 2;
      int col = (c & 3) << 3;
      copy16_g2l(&As[buf][row * 32 + col],
                 &A[(size_t)(m0 + row) * K + k0 + col]);
      copy16_g2l(&Bs[buf][row * 32 + col],
                 &Bt[(size_t)(n0 + row) * K + k0 + col]);
    }
  };

  v8f acc[2][2] = {};
  stage(0, 0);
  int cur = 0;

  for (int k0 = 0; k0 < K; k0 += 32) {
    const bool more = (k0 + 32) < K;
    if (more) {
      stage(cur ^ 1, k0 + 32);     // prefetch next tile behind compute
      async_wait_le<4>();          // in-order: <=4 outstanding => cur tile done
    } else {
      async_wait_le<0>();
    }
    __syncthreads();               // cur tile visible to all waves

    v16bf af[2], bf[2];
    af[0] = load_frag_rm(&As[cur][0], wm * 32 + 0,  32);
    af[1] = load_frag_rm(&As[cur][0], wm * 32 + 16, 32);
    bf[0] = load_frag_rm(&Bs[cur][0], wn * 32 + 0,  32);
    bf[1] = load_frag_rm(&Bs[cur][0], wn * 32 + 16, 32);
#pragma unroll
    for (int mi = 0; mi < 2; ++mi)
#pragma unroll
      for (int ni = 0; ni < 2; ++ni)
        acc[mi][ni] = wmma_bf16(af[mi], bf[ni], acc[mi][ni]);

    __syncthreads();               // all reads done before cur is overwritten
    cur ^= 1;
  }

  const int hf = lane >> 4;
  const int nc = lane & 15;
#pragma unroll
  for (int mi = 0; mi < 2; ++mi)
#pragma unroll
    for (int ni = 0; ni < 2; ++ni)
#pragma unroll
      for (int r = 0; r < 8; ++r) {
        int m = m0 + wm * 32 + mi * 16 + r + hf * 8;
        int n = n0 + wn * 32 + ni * 16 + nc;
        float val = acc[mi][ni][r];
        if (FINAL) Of[(size_t)m * N + n] = val + bias[n];
        else       Obf[(size_t)m * N + n] = f2bf(val);
      }
}

// ---------------------------------------------------------------------------
// Flash attention: 1 wave per (b, h, 16 query rows).  Online softmax.
// q,k stored [B,T,H*D] row-major (bf16); vT stored [B,H,D,T] (bf16).
// Output ao [B,T,H*D] bf16 (concatenated heads).
// All K and V fragments of an s-block are issued as one load clause up
// front so global latency overlaps the WMMA pipe and softmax VALU work.
// ---------------------------------------------------------------------------
__global__ __launch_bounds__(32)
void attn_kernel(const __bf16* __restrict__ q,
                 const __bf16* __restrict__ k,
                 const __bf16* __restrict__ vT,
                 __bf16* __restrict__ ao) {
  __shared__ alignas(16) __bf16 p_lds[16 * 32];

  const int t0 = blockIdx.x * 16;
  const int h  = blockIdx.y;
  const int b  = blockIdx.z;
  const int lane = threadIdx.x & 31;
  const int hf = lane >> 4;
  const int nc = lane & 15;

  const __bf16* qb = q  + (size_t)b * TT * CC + h * DD;          // row stride CC
  const __bf16* kb = k  + (size_t)b * TT * CC + h * DD;          // row stride CC
  const __bf16* vb = vT + ((size_t)b * HH + h) * DD * TT;        // [D][T], stride TT

  // q fragments for the full K=64 (two K-chunks of 32), kept in registers.
  v16bf qf[2];
#pragma unroll
  for (int kc = 0; kc < 2; ++kc) qf[kc] = load_frag_rm(qb + kc * 32, t0, CC);

  v8f o[4] = {};
  float m_i[8], l_i[8];
#pragma unroll
  for (int r = 0; r < 8; ++r) { m_i[r] = -__builtin_inff(); l_i[r] = 0.0f; }

  const float sc = 0.125f;          // 1/sqrt(64)
  const int smax = t0 + 16;         // exclusive bound on key index (causal)

  for (int s0 = 0; s0 < smax; s0 += 32) {
    // ---- issue ALL independent global loads for this s-block up front ----
    v16bf kf[2][2];                 // [s-subtile][K-chunk]
#pragma unroll
    for (int st = 0; st < 2; ++st)
#pragma unroll
      for (int kc = 0; kc < 2; ++kc)
        kf[st][kc] = load_frag_rm(kb + kc * 32, s0 + st * 16, CC);
    v16bf vf[4];                    // v fragments, independent of scores
#pragma unroll
    for (int dt = 0; dt < 4; ++dt)
      vf[dt] = load_frag_rm(vb + s0, dt * 16, TT);

    // prefetch next s-block's K/V into cache behind this block's compute
    if (s0 + 32 < smax) {
      __builtin_prefetch(kb + (size_t)(s0 + 32 + nc) * CC, 0, 1);
      __builtin_prefetch(kb + (size_t)(s0 + 48 + nc) * CC, 0, 1);
      __builtin_prefetch(vb + (size_t)(nc * 4 + hf) * TT + s0 + 32, 0, 1);
    }

    // scores tile 16x32 = two 16x16 WMMA outputs, each accumulated over K=64
    v8f s_t[2] = {};
#pragma unroll
    for (int st = 0; st < 2; ++st)
#pragma unroll
      for (int kc = 0; kc < 2; ++kc)
        s_t[st] = wmma_bf16(qf[kc], kf[st][kc], s_t[st]);

#pragma unroll
    for (int r = 0; r < 8; ++r) {
      const int trow = t0 + r + hf * 8;
      float a0 = s_t[0][r] * sc;
      float a1 = s_t[1][r] * sc;
      if (s0 + nc > trow)      a0 = -__builtin_inff();   // causal mask
      if (s0 + 16 + nc > trow) a1 = -__builtin_inff();

      float rowm = fmaxf(a0, a1);
#pragma unroll
      for (int off = 8; off >= 1; off >>= 1)
        rowm = fmaxf(rowm, __shfl_xor(rowm, off, 32));   // 16-lane butterfly
      float mnew = fmaxf(m_i[r], rowm);

      float p0 = __expf(a0 - mnew);
      float p1 = __expf(a1 - mnew);
      float rs = p0 + p1;
#pragma unroll
      for (int off = 8; off >= 1; off >>= 1)
        rs += __shfl_xor(rs, off, 32);

      float corr = __expf(m_i[r] - mnew);
      l_i[r] = l_i[r] * corr + rs;
      m_i[r] = mnew;
#pragma unroll
      for (int dt = 0; dt < 4; ++dt) o[dt][r] = o[dt][r] * corr;

      // C-layout -> LDS (row-major 16x32) for the p@v A-fragment
      p_lds[(r + hf * 8) * 32 + nc]      = f2bf(p0);
      p_lds[(r + hf * 8) * 32 + nc + 16] = f2bf(p1);
    }
    __syncthreads();

    v16bf pf = load_frag_rm(p_lds, 0, 32);
#pragma unroll
    for (int dt = 0; dt < 4; ++dt)
      o[dt] = wmma_bf16(pf, vf[dt], o[dt]);
    __syncthreads();
  }

  // epilogue: divide by row sums, store bf16 concat-head output
  __bf16* aob = ao + (size_t)b * TT * CC + h * DD;
#pragma unroll
  for (int r = 0; r < 8; ++r) {
    float inv = 1.0f / l_i[r];
    int trow = t0 + r + hf * 8;
#pragma unroll
    for (int dt = 0; dt < 4; ++dt)
      aob[(size_t)trow * CC + dt * 16 + nc] = f2bf(o[dt][r] * inv);
  }
}

// ---------------------------------------------------------------------------
// Host-side launch
// ---------------------------------------------------------------------------
extern "C" void kernel_launch(void* const* d_in, const int* in_sizes, int n_in,
                              void* d_out, int out_size, void* d_ws, size_t ws_size,
                              hipStream_t stream) {
  (void)in_sizes; (void)n_in; (void)out_size; (void)ws_size;
  const float* x  = (const float*)d_in[0];   // [B,T,C]
  const float* Wq = (const float*)d_in[1];   // [H,C,D]
  const float* Wk = (const float*)d_in[2];
  const float* Wv = (const float*)d_in[3];
  const float* Wp = (const float*)d_in[4];   // [H*D,C]
  const float* bp = (const float*)d_in[5];   // [C]

  const size_t M  = (size_t)BB * TT;         // 4096 token rows
  const size_t KC = CC;                      // 1024

  char* ws = (char*)d_ws;
  size_t off = 0;
  auto alloc = [&](size_t bytes) {
    char* p = ws + off;
    off += (bytes + 255) & ~(size_t)255;
    return p;
  };
  __bf16* xb  = (__bf16*)alloc(M * CC * 2);                 // x bf16
  __bf16* wqT = (__bf16*)alloc((size_t)HH * DD * CC * 2);   // [H*D, C]  (Bt)
  __bf16* wkT = (__bf16*)alloc((size_t)HH * DD * CC * 2);
  __bf16* wvT = (__bf16*)alloc((size_t)HH * DD * CC * 2);
  __bf16* wpT = (__bf16*)alloc((size_t)CC * HH * DD * 2);   // [C, H*D]  (Bt)
  __bf16* qb  = (__bf16*)alloc(M * CC * 2);                 // [B,T,H*D]
  __bf16* kb  = (__bf16*)alloc(M * CC * 2);
  __bf16* vb  = (__bf16*)alloc(M * CC * 2);
  __bf16* vTb = (__bf16*)alloc(M * CC * 2);                 // [B,H,D,T]
  __bf16* aob = (__bf16*)alloc(M * CC * 2);                 // [B,T,H*D]

  // 1) convert x, transpose-convert weights
  {
    int n = (int)(M * CC);
    cvt_f32_bf16_kernel<<<dim3((n + 255) / 256), dim3(256), 0, stream>>>(x, xb, n);
  }
  {
    int n = HH * CC * DD;  // 1M
    dim3 g((n + 255) / 256), blk(256);
    transpose_bf16_kernel<float><<<g, blk, 0, stream>>>(Wq, wqT, HH, CC, DD); // [h][d][c]
    transpose_bf16_kernel<float><<<g, blk, 0, stream>>>(Wk, wkT, HH, CC, DD);
    transpose_bf16_kernel<float><<<g, blk, 0, stream>>>(Wv, wvT, HH, CC, DD);
    transpose_bf16_kernel<float><<<g, blk, 0, stream>>>(Wp, wpT, 1, HH * DD, CC); // [c][hd]
  }

  // 2) QKV projections: [4096,1024] x [1024,1024]
  {
    dim3 g(CC / 64, (unsigned)(M / 64)), blk(128);
    gemm_bt_kernel<false><<<g, blk, 0, stream>>>(xb, wqT, qb, nullptr, nullptr,
                                                 (int)M, CC, (int)KC);
    gemm_bt_kernel<false><<<g, blk, 0, stream>>>(xb, wkT, kb, nullptr, nullptr,
                                                 (int)M, CC, (int)KC);
    gemm_bt_kernel<false><<<g, blk, 0, stream>>>(xb, wvT, vb, nullptr, nullptr,
                                                 (int)M, CC, (int)KC);
  }

  // 3) transpose v -> [B,H,D,T] so p@v B-fragments are contiguous
  {
    int n = (int)(M * CC);
    transpose_bf16_kernel<__bf16><<<dim3((n + 255) / 256), dim3(256), 0, stream>>>(
        vb, vTb, BB, TT, HH * DD);
  }

  // 4) causal flash attention
  {
    dim3 g(TT / 16, HH, BB), blk(32);
    attn_kernel<<<g, blk, 0, stream>>>(qb, kb, vTb, aob);
  }

  // 5) output projection + bias -> f32 d_out
  {
    dim3 g(CC / 64, (unsigned)(M / 64)), blk(128);
    gemm_bt_kernel<true><<<g, blk, 0, stream>>>(aob, wpT, nullptr, (float*)d_out,
                                                bp, (int)M, CC, HH * DD);
  }
}
